// MildenhallNeRF_59150289600925
// MI455X (gfx1250) — matrix operand order repaired and checked by reading, not compiled, not run on hardware
//
#include <hip/hip_runtime.h>
#include <hip/hip_bf16.h>
#include <math.h>

typedef __attribute__((ext_vector_type(16))) _Float16 v16h;
typedef __attribute__((ext_vector_type(8)))  float    v8f;

namespace {

constexpr int kT = 16384;   // hash table entries per level (T)
constexpr int kWaves = 8;   // waves per 256-thread block (wave32)

// NS[l] = int(16 * (512/16)^(l/15)) — matches numpy float64 evaluation.
constexpr int c_NS[16] = {16, 20, 25, 32, 40, 50, 64, 80,
                          101, 128, 161, 203, 256, 322, 406, 512};

union Frag {
  v16h v;
  _Float16 h[16];
  unsigned u[8];
};

__device__ inline v8f wmma32(v16h a, v16h b, v8f c) {
  return __builtin_amdgcn_wmma_f32_16x16x32_f16(false, a, false, b, (short)0, c,
                                                false, false);
}

// Opaque value: blocks LICM/CSE so loop-invariant weight fragments are
// re-loaded from LDS each tile instead of being pinned in ~160 VGPRs.
// Zero instructions emitted (empty asm), just pins x to a VGPR.
__device__ inline int opq(int x) {
  asm volatile("" : "+v"(x));
  return x;
}

// B fragment (K=32 x N=16 f16) for v_wmma_f32_16x16x32_f16.
// Per CDNA5 ISA B layout: lane n = lane&15 holds column n; lane group g carries
// K blocks {kbase+8g..+7, kbase+16+8g..+7} (same order as the A fragment), VGPR i
// packs a K pair. Weights staged in LDS transposed: wt[n*Kstride + k].
__device__ inline v16h load_bfrag(const _Float16* wt, int Kstride, int n,
                                  int kbase, int g) {
  Frag f;
  const unsigned* p = (const unsigned*)(wt + opq(n) * Kstride + kbase);
#pragma unroll
  for (int i = 0; i < 8; ++i)
    f.u[i] = p[(((i >= 4) ? 16 : 0) + g * 8 + (i & 3) * 2) >> 1];
  return f.v;
}

// A fragment (M=16 x K=32 f16): lane m holds row m; lane group g owns K blocks
// {8g..8g+7, 16+8g..16+8g+7}; VGPR i packs pairs per the ISA A-layout table.
// sc is a row-major [16][S] f16 LDS buffer.
__device__ inline v16h load_afrag(const _Float16* sc, int S, int m, int kbase,
                                  int g) {
  Frag f;
  const unsigned* p = (const unsigned*)(sc + m * S + kbase);
#pragma unroll
  for (int i = 0; i < 8; ++i) {
    const int kk = ((i >= 4) ? 16 : 0) + g * 8 + (i & 3) * 2;
    f.u[i] = p[kk >> 1];
  }
  return f.v;
}

__device__ inline float sigmoidf(float x) { return 1.0f / (1.0f + __expf(-x)); }

struct __align__(16) SMem {
  _Float16 dW0t[64 * 32];  // transposed dW0 [32x64] -> [n][k]
  _Float16 dW1t[16 * 64];  // transposed dW1 [64x16] -> [n][k]
  _Float16 cW0t[64 * 32];  // transposed cW0 [19x64] -> [n][k], k>=19 zero
  _Float16 cW1t[64 * 64];  // transposed cW1 [64x64] -> [n][k]
  _Float16 cW2t[16 * 64];  // transposed cW2 [64x3]  -> [n][k], n>=3 zero
  float db0[64];
  float db1[16];
  float cb0[64];
  float cb1[64];
  float cb2[16];
  _Float16 scratch[kWaves][16 * 64];  // per-wave activation transpose buffer
  float outstage[kWaves][16 * 4];     // per-wave output staging
};

}  // namespace

__global__ __launch_bounds__(256) void nerf_fused_kernel(
    const float* __restrict__ X,    // [B,6]: xyz + view
    const float* __restrict__ EMB,  // [16][16384][2]
    const float* __restrict__ dW0, const float* __restrict__ db0,
    const float* __restrict__ dW1, const float* __restrict__ db1,
    const float* __restrict__ cW0, const float* __restrict__ cb0,
    const float* __restrict__ cW1, const float* __restrict__ cb1,
    const float* __restrict__ cW2, const float* __restrict__ cb2,
    float* __restrict__ OUT,  // [B,4]: density, rgb
    int numTiles) {
  __shared__ SMem s;
  const int tid = threadIdx.x;

  // ---- Stage weights (f16, transposed) + biases into LDS (once/block) ----
  for (int i = tid; i < 64 * 32; i += 256) {
    const int n = i >> 5, k = i & 31;
    s.dW0t[i] = (_Float16)dW0[k * 64 + n];
  }
  for (int i = tid; i < 16 * 64; i += 256) {
    const int n = i >> 6, k = i & 63;
    s.dW1t[i] = (_Float16)dW1[k * 16 + n];
  }
  for (int i = tid; i < 64 * 32; i += 256) {
    const int n = i >> 5, k = i & 31;
    s.cW0t[i] = (k < 19) ? (_Float16)cW0[k * 64 + n] : (_Float16)0.f;
  }
  for (int i = tid; i < 64 * 64; i += 256) {
    const int n = i >> 6, k = i & 63;
    s.cW1t[i] = (_Float16)cW1[k * 64 + n];
  }
  for (int i = tid; i < 16 * 64; i += 256) {
    const int n = i >> 6, k = i & 63;
    s.cW2t[i] = (n < 3) ? (_Float16)cW2[k * 3 + n] : (_Float16)0.f;
  }
  if (tid < 64) s.db0[tid] = db0[tid];
  if (tid < 16) s.db1[tid] = db1[tid];
  if (tid < 64) s.cb0[tid] = cb0[tid];
  if (tid < 64) s.cb1[tid] = cb1[tid];
  if (tid < 16) s.cb2[tid] = (tid < 3) ? cb2[tid] : 0.f;
  __syncthreads();

  const int wv = tid >> 5;
  const int lane = tid & 31;
  const int m = lane & 15;  // point row within the 16-point tile
  const int g = lane >> 4;  // lane group
  _Float16* sc = s.scratch[wv];
  float* osg = s.outstage[wv];
  const float2* emb2 = (const float2*)EMB;

  const int gw = blockIdx.x * kWaves + wv;
  const int gwStride = gridDim.x * kWaves;

  for (int t = gw; t < numTiles; t += gwStride) {
    const int pbase = t << 4;
    const float* xp = X + (size_t)(pbase + m) * 6;
    const float cx = fminf(fmaxf((xp[0] + 5.f) * 0.1f, 0.f), 1.f - 1e-6f);
    const float cy = fminf(fmaxf((xp[1] + 5.f) * 0.1f, 0.f), 1.f - 1e-6f);
    const float cz = fminf(fmaxf((xp[2] + 5.f) * 0.1f, 0.f), 1.f - 1e-6f);

    // Prefetch next tile's input rows (global_prefetch_b8).
    {
      const int tn = t + gwStride;
      if (tn < numTiles) {
        __builtin_prefetch(X + (size_t)((tn << 4) + m) * 6, 0, 0);
      }
    }

    // ---- Hash-grid encode: builds the 16x32 f16 A fragment in registers.
    // Lane group g owns levels {4g..4g+3, 8+4g..8+4g+3}; feature K = 2*level.
    Frag fa;
#pragma unroll
    for (int i = 0; i < 8; ++i) {
      const int lvlBase = ((i >= 4) ? 8 : 0) + (i & 3);
      const int nl = g ? c_NS[lvlBase + 4] : c_NS[lvlBase];
      const int lvl = lvlBase + 4 * g;
      const float xl0 = cx * (float)nl;
      const float xl1 = cy * (float)nl;
      const float xl2 = cz * (float)nl;
      const float f0 = floorf(xl0), f1 = floorf(xl1), f2 = floorf(xl2);
      const float w0 = xl0 - f0, w1 = xl1 - f1, w2 = xl2 - f2;
      const int v0 = (int)f0, v1 = (int)f1, v2 = (int)f2;
      const int n1 = nl + 1;
      const bool dense = (n1 * n1 * n1 <= kT);
      float acc0 = 0.f, acc1 = 0.f;
#pragma unroll
      for (int cor = 0; cor < 8; ++cor) {
        const int o0 = (cor >> 2) & 1, o1 = (cor >> 1) & 1, o2 = cor & 1;
        unsigned idx;
        if (dense) {
          idx = (unsigned)((v0 + o0) * n1 * n1 + (v1 + o1) * n1 + (v2 + o2));
        } else {
          idx = (((unsigned)(v0 + o0)) * 1u ^
                 ((unsigned)(v1 + o1)) * 2654435761u ^
                 ((unsigned)(v2 + o2)) * 805459861u) &
                (unsigned)(kT - 1);
        }
        const float2 e = emb2[(size_t)lvl * kT + idx];  // global_load_b64, L2-hot
        const float w = (o0 ? w0 : 1.f - w0) * (o1 ? w1 : 1.f - w1) *
                        (o2 ? w2 : 1.f - w2);
        acc0 = fmaf(e.x, w, acc0);
        acc1 = fmaf(e.y, w, acc1);
      }
      fa.h[2 * i] = (_Float16)acc0;
      fa.h[2 * i + 1] = (_Float16)acc1;
    }

    // ---- Stage 1: h = relu(feats[16x32] @ dW0[32x64] + db0) -> scratch f16
#pragma unroll
    for (int nt = 0; nt < 4; ++nt) {
      const v16h b = load_bfrag(s.dW0t, 32, nt * 16 + m, 0, g);
      v8f c = {};
      c = wmma32(fa.v, b, c);
      const float bias = s.db0[nt * 16 + m];
#pragma unroll
      for (int r = 0; r < 8; ++r) {
        sc[(r + 8 * g) * 64 + nt * 16 + m] = (_Float16)fmaxf(c[r] + bias, 0.f);
      }
    }

    // ---- Stage 2: h2 = h[16x64] @ dW1[64x16] + db1 (K=64 -> chained WMMA)
    {
      const v16h alo = load_afrag(sc, 64, m, 0, g);
      const v16h ahi = load_afrag(sc, 64, m, 32, g);
      const v16h blo = load_bfrag(s.dW1t, 64, m, 0, g);
      const v16h bhi = load_bfrag(s.dW1t, 64, m, 32, g);
      v8f c = {};
      c = wmma32(alo, blo, c);
      c = wmma32(ahi, bhi, c);
      const float bias = s.db1[m];
#pragma unroll
      for (int r = 0; r < 8; ++r) {
        const int row = r + 8 * g;
        const float vv = c[r] + bias;
        const float den = sigmoidf(vv);
        const float act = (m == 0) ? den : fmaxf(vv, 0.f);
        sc[row * 32 + m] = (_Float16)act;  // col 0 sigmoid, cols 1..15 relu
        if (m == 0) osg[row * 4 + 0] = den;
      }
      if (g == 0) {  // append view dirs (cols 16..18) + zero pad to K=32
        const float* xv = X + (size_t)(pbase + m) * 6 + 3;
        sc[m * 32 + 16] = (_Float16)xv[0];
        sc[m * 32 + 17] = (_Float16)xv[1];
        sc[m * 32 + 18] = (_Float16)xv[2];
#pragma unroll
        for (int j = 19; j < 32; ++j) sc[m * 32 + j] = (_Float16)0.f;
      }
    }

    // ---- Stage 3: c1 = relu(cin[16x32] @ cW0[32x64] + cb0)
    {
      const v16h a = load_afrag(sc, 32, m, 0, g);
      v8f c[4];
#pragma unroll
      for (int nt = 0; nt < 4; ++nt) {
        const v16h b = load_bfrag(s.cW0t, 32, nt * 16 + m, 0, g);
        v8f z = {};
        c[nt] = wmma32(a, b, z);
      }
#pragma unroll
      for (int nt = 0; nt < 4; ++nt) {
        const float bias = s.cb0[nt * 16 + m];
#pragma unroll
        for (int r = 0; r < 8; ++r) {
          sc[(r + 8 * g) * 64 + nt * 16 + m] =
              (_Float16)fmaxf(c[nt][r] + bias, 0.f);
        }
      }
    }

    // ---- Stage 4: c2 = relu(c1[16x64] @ cW1[64x64] + cb1)
    {
      const v16h alo = load_afrag(sc, 64, m, 0, g);
      const v16h ahi = load_afrag(sc, 64, m, 32, g);
      v8f c[4];
#pragma unroll
      for (int nt = 0; nt < 4; ++nt) {
        const v16h blo = load_bfrag(s.cW1t, 64, nt * 16 + m, 0, g);
        const v16h bhi = load_bfrag(s.cW1t, 64, nt * 16 + m, 32, g);
        v8f z = {};
        z = wmma32(alo, blo, z);
        c[nt] = wmma32(ahi, bhi, z);
      }
#pragma unroll
      for (int nt = 0; nt < 4; ++nt) {
        const float bias = s.cb1[nt * 16 + m];
#pragma unroll
        for (int r = 0; r < 8; ++r) {
          sc[(r + 8 * g) * 64 + nt * 16 + m] =
              (_Float16)fmaxf(c[nt][r] + bias, 0.f);
        }
      }
    }

    // ---- Stage 5: color = sigmoid(c2[16x64] @ cW2[64x16pad] + cb2)
    {
      const v16h alo = load_afrag(sc, 64, m, 0, g);
      const v16h ahi = load_afrag(sc, 64, m, 32, g);
      const v16h blo = load_bfrag(s.cW2t, 64, m, 0, g);
      const v16h bhi = load_bfrag(s.cW2t, 64, m, 32, g);
      v8f c = {};
      c = wmma32(alo, blo, c);
      c = wmma32(ahi, bhi, c);
      if (m < 3) {
        const float bias = s.cb2[m];
#pragma unroll
        for (int r = 0; r < 8; ++r) {
          osg[(r + 8 * g) * 4 + 1 + m] = sigmoidf(c[r] + bias);
        }
      }
      if (g == 0) {  // one b128 store per point: [density, r, g, b]
        const float4 o = *(const float4*)&osg[m * 4];
        *(float4*)(OUT + (size_t)(pbase + m) * 4) = o;
      }
    }
  }
}

extern "C" void kernel_launch(void* const* d_in, const int* in_sizes, int n_in,
                              void* d_out, int out_size, void* d_ws,
                              size_t ws_size, hipStream_t stream) {
  (void)n_in; (void)out_size; (void)d_ws; (void)ws_size;
  const float* X   = (const float*)d_in[0];
  const float* EMB = (const float*)d_in[1];
  const float* dW0 = (const float*)d_in[2];
  const float* db0 = (const float*)d_in[3];
  const float* dW1 = (const float*)d_in[4];
  const float* db1 = (const float*)d_in[5];
  const float* cW0 = (const float*)d_in[6];
  const float* cb0 = (const float*)d_in[7];
  const float* cW1 = (const float*)d_in[8];
  const float* cb1 = (const float*)d_in[9];
  const float* cW2 = (const float*)d_in[10];
  const float* cb2 = (const float*)d_in[11];
  float* OUT = (float*)d_out;

  const int Bpts = in_sizes[0] / 6;
  const int numTiles = Bpts / 16;  // B = 2^21 -> 131072 tiles of 16 points

  int blocks = (numTiles + kWaves - 1) / kWaves;
  if (blocks > 2048) blocks = 2048;  // persistent waves, ~8 tiles per wave
  if (blocks < 1) blocks = 1;

  nerf_fused_kernel<<<blocks, 256, 0, stream>>>(
      X, EMB, dW0, db0, dW1, db1, cW0, cb0, cW1, cb1, cW2, cb2, OUT, numTiles);
}
